// SpatioTemporal_dot_80951543595348
// MI455X (gfx1250) — compile-verified, matrix operand order checked by reading, not run on hardware
//
#include <hip/hip_runtime.h>
#include <hip/hip_bf16.h>
#include <math.h>

typedef __attribute__((ext_vector_type(16))) _Float16 v16h;
typedef __attribute__((ext_vector_type(8)))  _Float16 v8h;
typedef __attribute__((ext_vector_type(8)))  float    v8f;

#define NTOT 5120              // T*H*W = 5*32*32
#define CIN  64
#define FOUT 10
#define NT_TILES (NTOT / 16)   // 320
#define NSPLIT 4               // split-K factor for both streaming passes

// ---------------------------------------------------------------------------
// Kernel 1: pointwise convs -> q16 [N][16] (f padded), k16 [N][16], v16 [16][N]
// ---------------------------------------------------------------------------
__global__ void st_qkv_kernel(const float* __restrict__ x1, const float* __restrict__ x2,
                              const float* __restrict__ w1, const float* __restrict__ b1,
                              const float* __restrict__ w2, const float* __restrict__ b2,
                              const float* __restrict__ w3, const float* __restrict__ b3,
                              _Float16* __restrict__ q16, _Float16* __restrict__ k16,
                              _Float16* __restrict__ v16) {
    int n = blockIdx.x * blockDim.x + threadIdx.x;
    if (n >= NTOT) return;
    int t = n >> 10;
    int s = n & 1023;
    const float* px1 = x1 + (size_t)t * CIN * 1024 + s;
    const float* px2 = x2 + (size_t)t * CIN * 1024 + s;

    float aq[FOUT], ak[FOUT], av[FOUT];
#pragma unroll
    for (int f = 0; f < FOUT; ++f) { aq[f] = 0.f; ak[f] = 0.f; av[f] = 0.f; }

    for (int c = 0; c < CIN; ++c) {
        float xa = px1[c * 1024];
        float xb = px2[c * 1024];
#pragma unroll
        for (int f = 0; f < FOUT; ++f) {
            aq[f] = fmaf(w1[f * CIN + c], xa, aq[f]);   // uniform w index -> s_load
            ak[f] = fmaf(w2[f * CIN + c], xb, ak[f]);
            av[f] = fmaf(w3[f * CIN + c], xa, av[f]);
        }
    }
#pragma unroll
    for (int f = 0; f < FOUT; ++f) {
        q16[(size_t)n * 16 + f] = (_Float16)(aq[f] + b1[f]);
        k16[(size_t)n * 16 + f] = (_Float16)(ak[f] + b2[f]);
        v16[(size_t)f * NTOT + n] = (_Float16)(av[f] + b3[f]);
    }
#pragma unroll
    for (int f = FOUT; f < 16; ++f) {      // zero padding for WMMA K / N dims
        q16[(size_t)n * 16 + f] = (_Float16)0.f;
        k16[(size_t)n * 16 + f] = (_Float16)0.f;
        v16[(size_t)f * NTOT + n] = (_Float16)0.f;
    }
}

// helpers: assemble v16h WMMA operands from 16-byte chunks
__device__ __forceinline__ v16h load_half_op(const _Float16* p) {
    v8h lo = *(const v8h*)p;
    v16h r = {};
#pragma unroll
    for (int i = 0; i < 8; ++i) r[i] = lo[i];
    return r;
}
__device__ __forceinline__ v16h load_full_op(const _Float16* plo, const _Float16* phi) {
    v8h lo = *(const v8h*)plo;
    v8h hi = *(const v8h*)phi;
    v16h r;
#pragma unroll
    for (int i = 0; i < 8; ++i) { r[i] = lo[i]; r[i + 8] = hi[i]; }
    return r;
}

// ---------------------------------------------------------------------------
// Kernel 2a: partial softmax column stats.  Wave = (m-tile, n-slice).
// 1280 waves, each streams NT_TILES/NSPLIT n-tiles through WMMA.
// ---------------------------------------------------------------------------
__global__ void st_colstats_part(const _Float16* __restrict__ q16,
                                 const _Float16* __restrict__ k16,
                                 float* __restrict__ pM, float* __restrict__ pS) {
    int lane = threadIdx.x & 31;
    int wave = (blockIdx.x * blockDim.x + threadIdx.x) >> 5;   // 0..NT_TILES*NSPLIT-1
    int mtile = wave >> 2;                                     // /NSPLIT
    int slice = wave & (NSPLIT - 1);
    int m_base = mtile * 16;
    int lm = lane & 15;
    int off = (lane < 16) ? 0 : 8;       // K-half split between lane halves

    // B operand: k columns m_base+lm, K = f (10 real, rest zero). Fixed per wave.
    v16h Bk = load_half_op(k16 + (size_t)(m_base + lm) * 16 + off);

    float runM = -INFINITY, runS = 0.f;
    int nt0 = slice * (NT_TILES / NSPLIT);
    for (int it = 0; it < NT_TILES / NSPLIT; ++it) {
        int nt = nt0 + it;
        v16h Aq = load_half_op(q16 + (size_t)(nt * 16 + lm) * 16 + off);
        v8f sc = {};
        sc = __builtin_amdgcn_wmma_f32_16x16x32_f16(false, Aq, false, Bk,
                                                    (short)0, sc, false, false);
        float tm = sc[0];
#pragma unroll
        for (int j = 1; j < 8; ++j) tm = fmaxf(tm, sc[j]);
        float newM = fmaxf(runM, tm);
        float add = 0.f;
#pragma unroll
        for (int j = 0; j < 8; ++j) add += __expf(sc[j] - newM);
        runS = runS * __expf(runM - newM) + add;   // exp(-inf)=0 on first iter
        runM = newM;
    }
    // lanes L and L+16 hold the same m column, disjoint n rows -> merge halves
    float oM = __shfl_xor(runM, 16);
    float oS = __shfl_xor(runS, 16);
    float M2 = fmaxf(runM, oM);
    float S2 = runS * __expf(runM - M2) + oS * __expf(oM - M2);
    if (lane < 16) {
        pM[(size_t)slice * NTOT + m_base + lm] = M2;
        pS[(size_t)slice * NTOT + m_base + lm] = S2;
    }
}

// Kernel 2b: merge NSPLIT partial (max,sum) pairs per column
__global__ void st_colstats_merge(const float* __restrict__ pM, const float* __restrict__ pS,
                                  float* __restrict__ colmax, float* __restrict__ colsum) {
    int m = blockIdx.x * blockDim.x + threadIdx.x;
    if (m >= NTOT) return;
    float M = pM[m], S = pS[m];
#pragma unroll
    for (int sl = 1; sl < NSPLIT; ++sl) {
        float Mi = pM[(size_t)sl * NTOT + m];
        float Si = pS[(size_t)sl * NTOT + m];
        float M2 = fmaxf(M, Mi);
        S = S * __expf(M - M2) + Si * __expf(Mi - M2);
        M = M2;
    }
    colmax[m] = M;
    colsum[m] = S;
}

// ---------------------------------------------------------------------------
// Kernel 3a: fused activations + output contraction, split over m.
// Wave = (n-tile, m-slice); 1280 waves; partial o -> po[slice][n][16].
// ---------------------------------------------------------------------------
__global__ void st_attn_out_part(const _Float16* __restrict__ q16,
                                 const _Float16* __restrict__ k16,
                                 const _Float16* __restrict__ v16,
                                 const float* __restrict__ colmax,
                                 const float* __restrict__ colsum,
                                 const float* __restrict__ actw,
                                 float* __restrict__ po) {
    __shared__ _Float16 lds_a[4][16 * 32];      // per-wave 16n x 32m f16 tile
    int lane = threadIdx.x & 31;
    int wib  = threadIdx.x >> 5;
    int waveId = blockIdx.x * 4 + wib;          // 0..NT_TILES*NSPLIT-1
    int ntile = waveId >> 2;                    // /NSPLIT
    int slice = waveId & (NSPLIT - 1);
    int n_base = ntile * 16;
    int lm = lane & 15;
    int off = (lane < 16) ? 0 : 8;
    int rowoff = (lane < 16) ? 0 : 8;           // D-tile: VGPR j -> row j (+8 hi half)
    float w0 = actw[0], w1 = actw[1];

    _Float16* aTile = lds_a[wib];

    // A operand (q rows of this n-tile) is loop-invariant: load once.
    v16h Aq = load_half_op(q16 + (size_t)(n_base + lm) * 16 + off);

    const int CHUNKS = NTOT / 32 / NSPLIT;      // 40 chunks of 32 m-columns
    int mc0 = slice * CHUNKS;

    v8f acc = {};
    for (int it = 0; it < CHUNKS; ++it) {
        int m0 = (mc0 + it) * 32;
#pragma unroll
        for (int h = 0; h < 2; ++h) {
            int mb = m0 + h * 16;
            v16h Bk = load_half_op(k16 + (size_t)(mb + lm) * 16 + off);
            v8f s = {};
            s = __builtin_amdgcn_wmma_f32_16x16x32_f16(false, Aq, false, Bk,
                                                       (short)0, s, false, false);
            // this lane owns column m = mb+lm, rows n_local = j+rowoff
            int mcol = mb + lm;
            float cm = colmax[mcol];
            float rs = w0 * __builtin_amdgcn_rcpf(colsum[mcol]);  // fold w0 (bug kept)
            int mloc = h * 16 + lm;
#pragma unroll
            for (int j = 0; j < 8; ++j) {
                float sv = s[j];
                float a = fmaf(w0, fmaxf(sv, 0.f),
                          fmaf(w1, __builtin_amdgcn_rcpf(1.f + __expf(-sv)),
                               rs * __expf(sv - cm)));
                aTile[(j + rowoff) * 32 + mloc] = (_Float16)a;
            }
        }
        // DS ops are in-order within a wave; wait for stores so cross-lane reads
        // below observe the full 16x32 tile.
        asm volatile("s_wait_dscnt 0" ::: "memory");

        // Re-read in WMMA A layout: row n = lm, K = m_local halves
        v16h Aa = load_full_op(aTile + lm * 32 + off,
                               aTile + lm * 32 + 16 + off);
        // B operand: v[f = lm, m0 + K]
        const _Float16* vr = v16 + (size_t)lm * NTOT + m0;
        v16h Bv = load_full_op(vr + off, vr + 16 + off);

        acc = __builtin_amdgcn_wmma_f32_16x16x32_f16(false, Aa, false, Bv,
                                                     (short)0, acc, false, false);
    }

    // acc[j] = partial o[n_base + j + rowoff, f = lm]
    float* pbase = po + ((size_t)slice * NTOT + n_base) * 16;
#pragma unroll
    for (int j = 0; j < 8; ++j) pbase[(j + rowoff) * 16 + lm] = acc[j];
}

// Kernel 3b: sum NSPLIT partial o buffers, scatter to [T,F,H,W]
__global__ void st_out_merge(const float* __restrict__ po, float* __restrict__ out) {
    int id = blockIdx.x * blockDim.x + threadIdx.x;   // n*16 + f
    if (id >= NTOT * 16) return;
    int f = id & 15;
    if (f >= FOUT) return;
    int n = id >> 4;
    float v = po[id];
#pragma unroll
    for (int sl = 1; sl < NSPLIT; ++sl) v += po[(size_t)sl * NTOT * 16 + id];
    int t = n >> 10, s = n & 1023;
    out[(size_t)(t * FOUT + f) * 1024 + s] = v;
}

// ---------------------------------------------------------------------------
extern "C" void kernel_launch(void* const* d_in, const int* in_sizes, int n_in,
                              void* d_out, int out_size, void* d_ws, size_t ws_size,
                              hipStream_t stream) {
    const float* input1 = (const float*)d_in[0];
    const float* input2 = (const float*)d_in[1];
    const float* actw   = (const float*)d_in[2];
    const float* c1w    = (const float*)d_in[3];
    const float* c1b    = (const float*)d_in[4];
    const float* c2w    = (const float*)d_in[5];
    const float* c2b    = (const float*)d_in[6];
    const float* g3w    = (const float*)d_in[7];
    const float* g3b    = (const float*)d_in[8];
    float* out = (float*)d_out;

    // workspace layout (~2.0 MB total; all hot data lives in L2)
    _Float16* q16 = (_Float16*)d_ws;              // 5120*16 halves
    _Float16* k16 = q16 + (size_t)NTOT * 16;
    _Float16* v16 = k16 + (size_t)NTOT * 16;      // 16*5120 halves
    float* colmax = (float*)(v16 + (size_t)16 * NTOT);
    float* colsum = colmax + NTOT;
    float* pM     = colsum + NTOT;                // [NSPLIT][NTOT]
    float* pS     = pM + (size_t)NSPLIT * NTOT;   // [NSPLIT][NTOT]
    float* po     = pS + (size_t)NSPLIT * NTOT;   // [NSPLIT][NTOT][16]

    // 1) q/k/v projections into f16 WMMA-friendly layouts
    st_qkv_kernel<<<(NTOT + 255) / 256, 256, 0, stream>>>(
        input1, input2, c1w, c1b, c2w, c2b, g3w, g3b, q16, k16, v16);

    // 2) softmax column stats: 1280 waves (4/block) + merge
    st_colstats_part<<<NT_TILES * NSPLIT / 4, 128, 0, stream>>>(q16, k16, pM, pS);
    st_colstats_merge<<<(NTOT + 255) / 256, 256, 0, stream>>>(pM, pS, colmax, colsum);

    // 3) fused activations + output contraction: 1280 waves (4/block) + merge
    st_attn_out_part<<<NT_TILES * NSPLIT / 4, 128, 0, stream>>>(
        q16, k16, v16, colmax, colsum, actw, po);
    st_out_merge<<<(NTOT * 16 + 255) / 256, 256, 0, stream>>>(po, out);
}
